// Bi_LSTM_CRF_61607010894281
// MI455X (gfx1250) — matrix operand order repaired and checked by reading, not compiled, not run on hardware
//
#include <hip/hip_runtime.h>

// Problem constants (from reference): V=50000, T=12, E=256, H=256, B=64, L=512
#define T_   12
#define E_   256
#define H_   256
#define B_   64
#define L_   512
#define G4H  1024              // 4*H
#define H2   512               // 2*H
#define ROWS (L_ * B_)         // 32768 = flattened [L, B]

typedef __attribute__((ext_vector_type(16))) __bf16 v16bf;
typedef __attribute__((ext_vector_type(8)))  float  v8f;

union TileU { v16bf v; uint4 q[2]; };

__device__ __forceinline__ unsigned short f2bf(float x) {
  unsigned int u = __float_as_uint(x);
  unsigned int r = u + 0x7FFFu + ((u >> 16) & 1u);   // round-to-nearest-even
  return (unsigned short)(r >> 16);
}

__device__ __forceinline__ float sigm(float x) { return 1.0f / (1.0f + __expf(-x)); }

__device__ __forceinline__ v8f wmma_bf16(v16bf a, v16bf b, v8f c) {
  // (neg_a, A, neg_b, B, c_mod, C, reuse_a, reuse_b)
  return __builtin_amdgcn_wmma_f32_16x16x32_bf16(false, a, false, b, (short)0, c,
                                                 false, false);
}

// A tile 16x32 (MxK), row-major source with leading dim ld (elements).
// lane m in [0,16): row m, K = {0..7, 16..23}; lane m+16: K = {8..15, 24..31}
__device__ __forceinline__ v16bf load_tileA(const unsigned short* base, int ld, int lane) {
  int m  = lane & 15;
  int ko = (lane >> 4) << 3;                 // 0 or 8
  const unsigned short* p = base + m * ld + ko;
  TileU t;
  t.q[0] = *(const uint4*)(p);               // K = ko .. ko+7
  t.q[1] = *(const uint4*)(p + 16);          // K = ko+16 .. ko+23
  return t.v;
}

// B tile 32x16 (KxN) sourced from row-major W [N, K] (column n of B = row n of W).
// lane n in [0,16): K = {0..7, 16..23}; lane n+16: K = {8..15, 24..31}.
__device__ __forceinline__ v16bf load_tileB(const unsigned short* base, int ld, int lane) {
  int n  = lane & 15;
  int ko = (lane >> 4) << 3;                 // 0 or 8
  const unsigned short* p = base + n * ld + ko;
  TileU t;
  t.q[0] = *(const uint4*)(p);
  t.q[1] = *(const uint4*)(p + 16);
  return t.v;
}

// Async global -> LDS copy of one 16-byte chunk (ASYNCcnt-tracked, CDNA5).
// lds_off is the wave-relative LDS byte address (dynamic-LDS base == 0 here).
__device__ __forceinline__ void async_g2l_b128(unsigned lds_off, const void* gaddr) {
  asm volatile("global_load_async_to_lds_b128 %0, %1, off"
               :: "v"(lds_off), "v"((unsigned long long)gaddr) : "memory");
}
__device__ __forceinline__ void wait_async0() {
  asm volatile("s_wait_asynccnt 0x0" ::: "memory");
}

// ---------------------------------------------------------------------------
// 1) Embedding gather + fp32 -> bf16.  X[l*B + b, :] = bf16(emb[sent[b,l], :])
// ---------------------------------------------------------------------------
__global__ void embed_kernel(const int* __restrict__ sentences,
                             const float* __restrict__ emb,
                             unsigned short* __restrict__ X) {
  int r = blockIdx.x;                       // r = l*64 + b
  int l = r >> 6, b = r & 63;
  int idx = sentences[b * L_ + l];
  const float* src = emb + (size_t)idx * E_;
  X[(size_t)r * E_ + threadIdx.x] = f2bf(src[threadIdx.x]);
}

// ---------------------------------------------------------------------------
// 2) Convert the four [1024,256] weight matrices to bf16 (row-major kept).
// ---------------------------------------------------------------------------
__global__ void convw_kernel(const float* __restrict__ w0, const float* __restrict__ w1,
                             const float* __restrict__ w2, const float* __restrict__ w3,
                             unsigned short* __restrict__ o0, unsigned short* __restrict__ o1,
                             unsigned short* __restrict__ o2, unsigned short* __restrict__ o3) {
  int g = blockIdx.x * blockDim.x + threadIdx.x;   // 4 * 262144 total
  int which = g >> 18;
  int i = g & 262143;
  const float* s; unsigned short* d;
  switch (which) {
    case 0: s = w0; d = o0; break;
    case 1: s = w1; d = o1; break;
    case 2: s = w2; d = o2; break;
    default: s = w3; d = o3; break;
  }
  d[i] = f2bf(s[i]);
}

// W_emit [12, 512] f32 -> padded bf16 [16, 512], rows 12..15 zero.
__global__ void convemit_kernel(const float* __restrict__ src,
                                unsigned short* __restrict__ dst) {
  int g = blockIdx.x * blockDim.x + threadIdx.x;   // 16*512
  int r = g >> 9, k = g & 511;
  dst[g] = (r < T_) ? f2bf(src[r * H2 + k]) : (unsigned short)0;
}

// ---------------------------------------------------------------------------
// 3) Input projection: Gpre[r, n] = sum_k X[r,k]*W[n,k] + bias[n]
//    Block = 256 thr (8 waves) owns 128 rows async-staged into 64 KB LDS.
//    Each wave hoists one N-tile's B k-chain into registers (64 VGPRs) and
//    sweeps the 8 row-tiles -> W read once per block, A from LDS.
// ---------------------------------------------------------------------------
__global__ __launch_bounds__(256) void gemm_inproj(const unsigned short* __restrict__ X,
                                                   const unsigned short* __restrict__ W,
                                                   const float* __restrict__ bias,
                                                   float* __restrict__ out) {
  extern __shared__ unsigned char smem[];
  unsigned short* As = (unsigned short*)smem;          // [128][256] bf16 = 64 KB
  int tid = threadIdx.x, wave = tid >> 5, lane = tid & 31;
  int row_base = blockIdx.x * 128;

  // async stage: 128 rows x 256 cols bf16 = 4096 x b128, direct global -> LDS
  {
    const uint4* src = (const uint4*)(X + (size_t)row_base * E_);
    for (int i = tid; i < 4096; i += 256)
      async_g2l_b128((unsigned)(i * 16), (const void*)(src + i));
    wait_async0();
  }
  __syncthreads();

  for (int i = 0; i < 8; ++i) {
    int tile_n = wave * 8 + i;                         // 0..63
    const unsigned short* Bb = W + (size_t)tile_n * 16 * E_;
    v16bf Bt[8];
#pragma unroll
    for (int ks = 0; ks < 8; ++ks) Bt[ks] = load_tileB(Bb + ks * 32, E_, lane);

    int col = tile_n * 16 + (lane & 15);
    float bv = bias[col];
    for (int rt = 0; rt < 8; ++rt) {
      const unsigned short* Ab = As + rt * 16 * E_;
      v8f acc = {};
#pragma unroll
      for (int ks = 0; ks < 8; ++ks) {
        v16bf a = load_tileA(Ab + ks * 32, E_, lane);  // LDS
        acc = wmma_bf16(a, Bt[ks], acc);
      }
      int row0 = row_base + rt * 16 + ((lane >> 4) ? 8 : 0);
#pragma unroll
      for (int j = 0; j < 8; ++j)
        out[(size_t)(row0 + j) * G4H + col] = acc[j] + bv;
    }
  }
}

// ---------------------------------------------------------------------------
// 4) Sequential LSTM, one persistent workgroup per direction (grid = 2).
//    LDS: g_buf [64][1024] f32 (256 KB) + h_buf [64][256] bf16 (32 KB).
//    Per step: acc <- Gpre[t] (WMMA C operand), acc += h_prev @ W_hh^T,
//    gate nonlinearity with c in registers, h -> LDS bf16 + hcat global bf16.
// ---------------------------------------------------------------------------
__global__ __launch_bounds__(1024) void lstm_seq(const unsigned short* __restrict__ Whh_f,
                                                 const unsigned short* __restrict__ Whh_b,
                                                 const float* __restrict__ Gpre_f,
                                                 const float* __restrict__ Gpre_b,
                                                 unsigned short* __restrict__ hcat) {
  extern __shared__ unsigned char smem[];
  float*          g_buf = (float*)smem;                              // 64*1024
  unsigned short* h_buf = (unsigned short*)(smem + B_ * G4H * 4);    // 64*256

  int dir = blockIdx.x;
  const unsigned short* Whh  = dir ? Whh_b  : Whh_f;
  const float*          Gpre = dir ? Gpre_b : Gpre_f;
  unsigned short*       hout = hcat + dir * H_;   // fwd: cols 0..255, bwd: 256..511

  int tid = threadIdx.x, wave = tid >> 5, lane = tid & 31;

  for (int i = tid; i < B_ * H_; i += 1024) h_buf[i] = 0;            // h0 = 0
  float c_reg[16];
#pragma unroll
  for (int i = 0; i < 16; ++i) c_reg[i] = 0.0f;                      // c0 = 0
  __syncthreads();

  for (int s = 0; s < L_; ++s) {
    int t = dir ? (L_ - 1 - s) : s;
    const float* gp = Gpre + (size_t)t * B_ * G4H;

    // prefetch next step's gate preactivations (256 KB) into cache
    if (s + 1 < L_) {
      int tn = dir ? (t - 1) : (t + 1);
      const float* gpn = Gpre + (size_t)tn * B_ * G4H;
      __builtin_prefetch(gpn + tid * 64, 0, 0);
      __builtin_prefetch(gpn + tid * 64 + 32, 0, 0);
    }

    // ---- recurrent GEMM: per wave, 2 N-tiles (B hoisted to regs) x 4 M-tiles
#pragma unroll
    for (int tn2 = 0; tn2 < 2; ++tn2) {
      int tile_n = wave * 2 + tn2;
      const unsigned short* Bb = Whh + (size_t)tile_n * 16 * H_;
      v16bf Bt[8];
#pragma unroll
      for (int ks = 0; ks < 8; ++ks) Bt[ks] = load_tileB(Bb + ks * 32, H_, lane);

      int col = tile_n * 16 + (lane & 15);
      for (int tm = 0; tm < 4; ++tm) {
        const unsigned short* Ab = h_buf + tm * 16 * H_;
        int brow0 = tm * 16 + ((lane >> 4) ? 8 : 0);
        v8f acc;
#pragma unroll
        for (int j = 0; j < 8; ++j)                    // C operand <- Gpre[t]
          acc[j] = gp[(size_t)(brow0 + j) * G4H + col];
#pragma unroll
        for (int ks = 0; ks < 8; ++ks) {
          v16bf a = load_tileA(Ab + ks * 32, H_, lane);   // LDS
          acc = wmma_bf16(a, Bt[ks], acc);
        }
#pragma unroll
        for (int j = 0; j < 8; ++j)
          g_buf[(brow0 + j) * G4H + col] = acc[j];
      }
    }
    __syncthreads();

    // ---- gates: torch order i, f, g, o ; 16 (b, j) cells per thread
#pragma unroll
    for (int i = 0; i < 16; ++i) {
      int cell = tid + (i << 10);
      int b = cell >> 8, j = cell & 255;
      float ig = g_buf[b * G4H + j];
      float fg = g_buf[b * G4H + 256 + j];
      float gg = g_buf[b * G4H + 512 + j];
      float og = g_buf[b * G4H + 768 + j];
      float c = sigm(fg) * c_reg[i] + sigm(ig) * tanhf(gg);
      float h = sigm(og) * tanhf(c);
      c_reg[i] = c;
      unsigned short hb = f2bf(h);
      h_buf[b * H_ + j] = hb;
      hout[((size_t)t * B_ + b) * H2 + j] = hb;
    }
    __syncthreads();
  }
}

// ---------------------------------------------------------------------------
// 5) Emit via WMMA: emit[r, t] = hcat[r,:] . Wem_bf[t,:] + b_emit[t]
//    hcat [32768, 512] bf16, Wem padded [16, 512] bf16; store cols 0..11.
//    Block = 256 thr (8 waves); one 16x16 output tile per wave.
// ---------------------------------------------------------------------------
__global__ __launch_bounds__(256) void emit_gemm(const unsigned short* __restrict__ hcat,
                                                 const unsigned short* __restrict__ Wem,
                                                 const float* __restrict__ b_emit,
                                                 float* __restrict__ emit) {
  int wave = threadIdx.x >> 5, lane = threadIdx.x & 31;
  int tile_r = blockIdx.x * 8 + wave;                  // 0..2047
  const unsigned short* Ab = hcat + (size_t)tile_r * 16 * H2;
  v8f acc = {};
#pragma unroll
  for (int ks = 0; ks < 16; ++ks) {                    // K = 512
    v16bf a = load_tileA(Ab + ks * 32, H2, lane);
    v16bf b = load_tileB(Wem + ks * 32, H2, lane);
    acc = wmma_bf16(a, b, acc);
  }
  int col = lane & 15;
  if (col < T_) {
    float bv = b_emit[col];
    int row0 = tile_r * 16 + ((lane >> 4) ? 8 : 0);
#pragma unroll
    for (int j = 0; j < 8; ++j)
      emit[(size_t)(row0 + j) * T_ + col] = acc[j] + bv;
  }
}

// ---------------------------------------------------------------------------
// 6) CRF NLL: forward logsumexp scan + gold score; one thread per batch elem.
// ---------------------------------------------------------------------------
__global__ void crf_kernel(const float* __restrict__ emit, const float* __restrict__ trans,
                           const int* __restrict__ tags, float* __restrict__ out) {
  __shared__ float tr[T_ * T_];
  __shared__ float red[B_];
  int b = threadIdx.x;
  for (int i = b; i < T_ * T_; i += B_) tr[i] = trans[i];
  __syncthreads();

  float alpha[T_];
#pragma unroll
  for (int j = 0; j < T_; ++j) alpha[j] = emit[(size_t)(0 * B_ + b) * T_ + j];

  for (int t = 1; t < L_; ++t) {
    const float* et = emit + (size_t)(t * B_ + b) * T_;
    float na[T_];
#pragma unroll
    for (int j = 0; j < T_; ++j) {
      float m = -3.0e38f;
#pragma unroll
      for (int i = 0; i < T_; ++i) m = fmaxf(m, alpha[i] + tr[i * T_ + j]);
      float s = 0.0f;
#pragma unroll
      for (int i = 0; i < T_; ++i) s += __expf(alpha[i] + tr[i * T_ + j] - m);
      na[j] = m + __logf(s) + et[j];
    }
#pragma unroll
    for (int j = 0; j < T_; ++j) alpha[j] = na[j];
  }

  float m = -3.0e38f;
#pragma unroll
  for (int j = 0; j < T_; ++j) m = fmaxf(m, alpha[j]);
  float s = 0.0f;
#pragma unroll
  for (int j = 0; j < T_; ++j) s += __expf(alpha[j] - m);
  float logZ = m + __logf(s);

  float eg = 0.0f;
  for (int t = 0; t < L_; ++t)
    eg += emit[(size_t)(t * B_ + b) * T_ + tags[b * L_ + t]];
  float tg = 0.0f;
  for (int t = 0; t < L_ - 1; ++t)
    tg += tr[tags[b * L_ + t] * T_ + tags[b * L_ + t + 1]];

  red[b] = logZ - eg - tg;
  __syncthreads();
  if (b == 0) {
    float acc = 0.0f;
    for (int i = 0; i < B_; ++i) acc += red[i];
    out[0] = acc / (float)B_;
  }
}

// ---------------------------------------------------------------------------
extern "C" void kernel_launch(void* const* d_in, const int* in_sizes, int n_in,
                              void* d_out, int out_size, void* d_ws, size_t ws_size,
                              hipStream_t stream) {
  const int*   sentences = (const int*)d_in[0];
  const int*   tags      = (const int*)d_in[1];
  const float* embedding = (const float*)d_in[2];
  const float* W_ih_f    = (const float*)d_in[3];
  const float* W_hh_f    = (const float*)d_in[4];
  const float* b_f       = (const float*)d_in[5];
  const float* W_ih_b    = (const float*)d_in[6];
  const float* W_hh_b    = (const float*)d_in[7];
  const float* b_b       = (const float*)d_in[8];
  const float* W_emit    = (const float*)d_in[9];
  const float* b_emit    = (const float*)d_in[10];
  const float* trans     = (const float*)d_in[11];
  float*       out       = (float*)d_out;

  unsigned char* ws = (unsigned char*)d_ws;
  size_t off = 0;
  unsigned short* Xbf   = (unsigned short*)(ws + off); off += (size_t)ROWS * E_ * 2;
  unsigned short* Wihf  = (unsigned short*)(ws + off); off += (size_t)G4H * E_ * 2;
  unsigned short* Wihb  = (unsigned short*)(ws + off); off += (size_t)G4H * E_ * 2;
  unsigned short* Whhf  = (unsigned short*)(ws + off); off += (size_t)G4H * H_ * 2;
  unsigned short* Whhb  = (unsigned short*)(ws + off); off += (size_t)G4H * H_ * 2;
  unsigned short* Wemb  = (unsigned short*)(ws + off); off += (size_t)16 * H2 * 2;
  float* Gpre_f = (float*)(ws + off); off += (size_t)ROWS * G4H * 4;
  float* Gpre_b = (float*)(ws + off); off += (size_t)ROWS * G4H * 4;
  unsigned short* hcat = (unsigned short*)(ws + off); off += (size_t)ROWS * H2 * 2;
  float* emitb  = (float*)(ws + off); off += (size_t)ROWS * T_ * 4;

  embed_kernel<<<ROWS, 256, 0, stream>>>(sentences, embedding, Xbf);
  convw_kernel<<<4096, 256, 0, stream>>>(W_ih_f, W_ih_b, W_hh_f, W_hh_b,
                                         Wihf, Wihb, Whhf, Whhb);
  convemit_kernel<<<32, 256, 0, stream>>>(W_emit, Wemb);

  (void)hipFuncSetAttribute((const void*)gemm_inproj,
                            hipFuncAttributeMaxDynamicSharedMemorySize, 65536);
  gemm_inproj<<<ROWS / 128, 256, 65536, stream>>>(Xbf, Wihf, b_f, Gpre_f);
  gemm_inproj<<<ROWS / 128, 256, 65536, stream>>>(Xbf, Wihb, b_b, Gpre_b);

  size_t lds_bytes = (size_t)B_ * G4H * 4 + (size_t)B_ * H_ * 2;  // 288 KB
  (void)hipFuncSetAttribute((const void*)lstm_seq,
                            hipFuncAttributeMaxDynamicSharedMemorySize,
                            (int)lds_bytes);
  lstm_seq<<<2, 1024, lds_bytes, stream>>>(Whhf, Whhb, Gpre_f, Gpre_b, hcat);

  emit_gemm<<<ROWS / 128, 256, 0, stream>>>(hcat, Wemb, b_emit, emitb);
  crf_kernel<<<1, B_, 0, stream>>>(emitb, trans, tags, out);
}
